// MetaGCRN_60593398612447
// MI455X (gfx1250) — matrix operand order repaired and checked by reading, not compiled, not run on hardware
//
#include <hip/hip_runtime.h>
#include <hip/hip_bf16.h>
#include <math.h>

typedef __bf16 bf16_t;
typedef __attribute__((ext_vector_type(16))) __bf16 v16bf;
typedef __attribute__((ext_vector_type(8)))  float  v8f;

#define NN 1024
#define BB 64
#define TT 12
#define HH 64
#define CP 72              // padded per-sample feature width (multiple of 8)
#define FF (BB * CP)       // 4608 columns for the S/T2 GEMMs
#define WKP 96             // weight GEMM K (zero-padded weight rows 72..95)

// ---------------- generic bf16 WMMA GEMM (all tiles full by construction) ----
// C(MxN) = A(MxK') @ B(K'xN); A,B bf16 row-major.
// REQUIREMENTS (guaranteed by caller): M%128==0, N%64==0, K%32==0,
// lda%8==0, ldb%8==0, pointers 16B aligned.  A may be logically narrower than
// K (lda < K) as long as the extra B rows are zero and A's over-read stays in
// valid, finite memory (guard-padded).
// MODE 0: Cf  = acc
// MODE 1: Cf += acc
// MODE 2: Cf  = sigmoid(Cf + acc + bias[n])
// MODE 3: Cf  = tanh   (Cf + acc + bias[n])
// MODE 4: Dbf = acc                (bf16 out)
// MODE 5: Dbf = 2*acc - I          (bf16 out, Chebyshev T2)
#define GBM 128
#define GBN 64
#define GBK 32

union FragU { v16bf v; uint4 q[2]; };

template <int MODE>
__global__ __launch_bounds__(256)
void gemm_bf16(const bf16_t* __restrict__ A, int lda,
               const bf16_t* __restrict__ B, int ldb,
               float* __restrict__ Cf, bf16_t* __restrict__ Dbf,
               const float* __restrict__ bias,
               int N, int K)
{
    __shared__ __align__(16) bf16_t As[GBM][GBK];
    __shared__ __align__(16) bf16_t Bs[GBN][GBK];   // stored transposed: Bs[n][k]

    const int tid  = threadIdx.x;
    const int lane = tid & 31;
    const int wave = tid >> 5;
    const int wm   = wave >> 1;        // 0..3
    const int wn   = wave & 1;         // 0..1
    const int l16  = lane & 15;
    const int lh8  = (lane >> 4) * 8;  // 0 or 8
    const int m0   = blockIdx.y * GBM;
    const int n0   = blockIdx.x * GBN;

    v8f acc[2][2];
#pragma unroll
    for (int i = 0; i < 2; ++i)
#pragma unroll
        for (int j = 0; j < 2; ++j)
#pragma unroll
            for (int r = 0; r < 8; ++r) acc[i][j][r] = 0.0f;

    // A staging: each thread loads 16 contiguous elements of one tile row
    const int ar = tid >> 1;          // tile row (0..127)
    const int ac = (tid & 1) * 16;    // col base (0 or 16)
    // B staging: each thread covers k rows {2kp,2kp+1} x 4 columns
    const int kp = tid >> 4;          // 0..15
    const int nb = (tid & 15) * 4;    // 0..60

    const bf16_t* pA = A + (size_t)(m0 + ar) * lda + ac;
    const bf16_t* pB = B + (size_t)(2 * kp) * ldb + n0 + nb;

    for (int k0 = 0; k0 < K; k0 += GBK) {
        // ---- stage A tile (vectorized, no bounds checks) ----
        *(uint4*)&As[ar][ac]     = *(const uint4*)(pA);
        *(uint4*)&As[ar][ac + 8] = *(const uint4*)(pA + 8);
        pA += GBK;
        // ---- stage B tile transposed into Bs[n][k] ----
        {
            uint2 r0 = *(const uint2*)(pB);
            uint2 r1 = *(const uint2*)(pB + ldb);
            pB += (size_t)GBK * ldb;
            const unsigned short* s0 = (const unsigned short*)&r0;
            const unsigned short* s1 = (const unsigned short*)&r1;
#pragma unroll
            for (int i = 0; i < 4; ++i) {
                unsigned int pk = (unsigned int)s0[i] | ((unsigned int)s1[i] << 16);
                *(unsigned int*)&Bs[nb + i][2 * kp] = pk;   // pairs (k, k+1)
            }
        }
        __syncthreads();

        FragU a[2], b[2];
#pragma unroll
        for (int i = 0; i < 2; ++i) {
            int row = wm * 32 + i * 16 + l16;
            // 16-bit A 16x32 layout: lanes<16 hold K[0..7]+K[16..23], lanes>=16 K[8..15]+K[24..31]
            a[i].q[0] = *(const uint4*)&As[row][lh8];
            a[i].q[1] = *(const uint4*)&As[row][lh8 + 16];
        }
#pragma unroll
        for (int j = 0; j < 2; ++j) {
            int col = wn * 32 + j * 16 + l16;
            // B 32x16: lanes<16 hold K[0..7]+K[16..23], lanes>=16 K[8..15]+K[24..31] of their column
            b[j].q[0] = *(const uint4*)&Bs[col][lh8];
            b[j].q[1] = *(const uint4*)&Bs[col][lh8 + 16];
        }
#pragma unroll
        for (int i = 0; i < 2; ++i)
#pragma unroll
            for (int j = 0; j < 2; ++j)
                acc[i][j] = __builtin_amdgcn_wmma_f32_16x16x32_bf16(
                    false, a[i].v, false, b[j].v, (short)0, acc[i][j], false, false);
        __syncthreads();
    }

#pragma unroll
    for (int i = 0; i < 2; ++i)
#pragma unroll
        for (int j = 0; j < 2; ++j)
#pragma unroll
            for (int r = 0; r < 8; ++r) {
                int m = m0 + wm * 32 + i * 16 + lh8 + r;   // f32 C/D: M = r + 8*(lane>=16)
                int n = n0 + wn * 32 + j * 16 + l16;       // N = lane&15
                size_t idx = (size_t)m * N + n;
                float v = acc[i][j][r];
                if (MODE == 0)      Cf[idx] = v;
                else if (MODE == 1) Cf[idx] += v;
                else if (MODE == 2) { float s = Cf[idx] + v + bias[n]; Cf[idx] = 1.0f / (1.0f + expf(-s)); }
                else if (MODE == 3) { float s = Cf[idx] + v + bias[n]; Cf[idx] = tanhf(s); }
                else if (MODE == 4) Dbf[idx] = (bf16_t)v;
                else                { float s = 2.0f * v - ((m == n) ? 1.0f : 0.0f); Dbf[idx] = (bf16_t)s; }
            }
}

// ---------------- support matrix: S = softmax(relu(emb @ emb^T), axis=1) -> bf16 ----------------
__global__ __launch_bounds__(256)
void row_softmax(const float* __restrict__ emb, int Cdim, bf16_t* __restrict__ S)
{
    __shared__ float srow[NN];
    __shared__ float red[256];
    const int n = blockIdx.x, tid = threadIdx.x;

    for (int m = tid; m < NN; m += 256) {
        float s = 0.0f;
        for (int c = 0; c < Cdim; ++c)
            s += emb[(size_t)n * Cdim + c] * emb[(size_t)m * Cdim + c];
        srow[m] = fmaxf(s, 0.0f);
    }
    __syncthreads();

    float mx = -1e30f;
    for (int m = tid; m < NN; m += 256) mx = fmaxf(mx, srow[m]);
    red[tid] = mx; __syncthreads();
    for (int s = 128; s > 0; s >>= 1) { if (tid < s) red[tid] = fmaxf(red[tid], red[tid + s]); __syncthreads(); }
    mx = red[0]; __syncthreads();

    float sum = 0.0f;
    for (int m = tid; m < NN; m += 256) { float e = expf(srow[m] - mx); srow[m] = e; sum += e; }
    red[tid] = sum; __syncthreads();
    for (int s = 128; s > 0; s >>= 1) { if (tid < s) red[tid] += red[tid + s]; __syncthreads(); }
    float inv = 1.0f / red[0];

    for (int m = tid; m < NN; m += 256) S[(size_t)n * NN + m] = (bf16_t)(srow[m] * inv);
}

// ---------------- small helpers ----------------
// Repack GRU weight (3 pieces of c rows) into zero-padded bf16 [3*WKP][ncols]
__global__ void repack_w(const float* __restrict__ W, int c, int ncols, bf16_t* __restrict__ Wp)
{
    int i = blockIdx.x * 256 + threadIdx.x;
    if (i >= 3 * WKP * ncols) return;
    int row = i / ncols, col = i - row * ncols;
    int p = row / WKP, r = row - p * WKP;
    Wp[i] = (r < c) ? (bf16_t)W[(size_t)(p * c + r) * ncols + col] : (bf16_t)0.0f;
}

__global__ void zero_f32(float* __restrict__ p, int n)
{
    int i = blockIdx.x * blockDim.x + threadIdx.x;
    if (i < n) p[i] = 0.0f;
}

// Xg layout: [n][b][CP] flattened -> row m=n*B+b of a (B*N x CP) matrix, and the
// (N x B*CP) RHS for S @ X simultaneously.  Tail columns are zero.
__global__ void build_x_enc(const float* __restrict__ x, int t,
                            const float* __restrict__ H, const float* __restrict__ ZR,
                            int use_z, bf16_t* __restrict__ Xg)
{
    int idx = blockIdx.x * 256 + threadIdx.x;   // 65536 = n*B+b
    int b = idx & 63, n = idx >> 6;
    size_t base = (size_t)idx * CP;
    Xg[base] = (bf16_t)x[(size_t)b * (TT * NN) + (size_t)t * NN + n];
    for (int j = 0; j < HH; ++j) {
        float hv = H[(size_t)idx * HH + j];
        if (use_z) hv *= ZR[(size_t)idx * 128 + j];
        Xg[base + 1 + j] = (bf16_t)hv;
    }
    for (int j = 1 + HH; j < CP; ++j) Xg[base + j] = (bf16_t)0.0f;
}

__global__ void build_x_dec(const float* __restrict__ go, const float* __restrict__ ycov, int t,
                            const float* __restrict__ H, const float* __restrict__ ZR,
                            int use_z, bf16_t* __restrict__ Xg)
{
    int idx = blockIdx.x * 256 + threadIdx.x;
    int b = idx & 63, n = idx >> 6;
    size_t base = (size_t)idx * CP;
    Xg[base]     = (bf16_t)go[(size_t)b * NN + n];
    Xg[base + 1] = (bf16_t)ycov[(size_t)b * (TT * NN) + (size_t)t * NN + n];
    for (int j = 0; j < HH; ++j) {
        float hv = H[(size_t)idx * HH + j];
        if (use_z) hv *= ZR[(size_t)idx * 128 + j];
        Xg[base + 2 + j] = (bf16_t)hv;
    }
    for (int j = 2 + HH; j < CP; ++j) Xg[base + j] = (bf16_t)0.0f;
}

__global__ void gru_combine(float* __restrict__ H, const float* __restrict__ ZR,
                            const float* __restrict__ HC)
{
    int i = blockIdx.x * 256 + threadIdx.x;     // 65536*64
    int m = i >> 6, j = i & 63;
    float r = ZR[(size_t)m * 128 + 64 + j];
    H[i] = r * H[i] + (1.0f - r) * HC[i];
}

// M2T[n][b*10+e] = sum_j H[n,b,j] * FC_E[j,e]   (decoder adaptive embeddings)
__global__ void dec_emb_kernel(const float* __restrict__ H, const float* __restrict__ FCE,
                               float* __restrict__ M2T)
{
    int idx = blockIdx.x * 256 + threadIdx.x;   // n*B+b
    int b = idx & 63, n = idx >> 6;
    for (int e = 0; e < 10; ++e) {
        float s = 0.0f;
        for (int j = 0; j < HH; ++j) s += H[(size_t)idx * HH + j] * FCE[j * 10 + e];
        M2T[(size_t)n * (BB * 10) + b * 10 + e] = s;
    }
}

__global__ void go_init(const float* __restrict__ x, float* __restrict__ go)
{
    int idx = blockIdx.x * 256 + threadIdx.x;   // b*N+n
    int b = idx >> 10, n = idx & 1023;
    go[idx] = x[(size_t)b * (TT * NN) + (size_t)(TT - 1) * NN + n];
}

__global__ void proj_kernel(const float* __restrict__ H, const float* __restrict__ pw,
                            const float* __restrict__ pb, float* __restrict__ go,
                            float* __restrict__ out, int t)
{
    int idx = blockIdx.x * 256 + threadIdx.x;   // b*N+n
    int b = idx >> 10, n = idx & 1023;
    int m = (n << 6) + b;
    float s = pb[0];
    for (int j = 0; j < HH; ++j) s += H[(size_t)m * HH + j] * pw[j];
    go[idx] = s;
    out[((size_t)b * TT + t) * NN + n] = s;
}

// ---------------- host orchestration ----------------
extern "C" void kernel_launch(void* const* d_in, const int* in_sizes, int n_in,
                              void* d_out, int out_size, void* d_ws, size_t ws_size,
                              hipStream_t stream)
{
    (void)in_sizes; (void)n_in; (void)out_size; (void)ws_size;

    const float* x    = (const float*)d_in[0];
    const float* ycov = (const float*)d_in[1];
    const float* emb  = (const float*)d_in[2];
    const float* FCE  = (const float*)d_in[3];
    const float* egw  = (const float*)d_in[4];
    const float* egb  = (const float*)d_in[5];
    const float* euw  = (const float*)d_in[6];
    const float* eub  = (const float*)d_in[7];
    const float* dgw  = (const float*)d_in[8];
    const float* dgb  = (const float*)d_in[9];
    const float* duw  = (const float*)d_in[10];
    const float* dub  = (const float*)d_in[11];
    const float* pw   = (const float*)d_in[12];
    const float* pb   = (const float*)d_in[13];
    float* out = (float*)d_out;

    char* wsp = (char*)d_ws;
    auto carve = [&](size_t bytes) -> void* {
        void* p = (void*)wsp;
        wsp += (bytes + 255) & ~(size_t)255;
        return p;
    };
    const int MROWS = BB * NN;            // 65536
    bf16_t* S    = (bf16_t*)carve((size_t)NN * NN * 2);
    bf16_t* T2   = (bf16_t*)carve((size_t)NN * NN * 2);
    bf16_t* WGE  = (bf16_t*)carve((size_t)3 * WKP * 128 * 2);
    bf16_t* WUE  = (bf16_t*)carve((size_t)3 * WKP * 64 * 2);
    bf16_t* WGD  = (bf16_t*)carve((size_t)3 * WKP * 128 * 2);
    bf16_t* WUD  = (bf16_t*)carve((size_t)3 * WKP * 64 * 2);
    bf16_t* Xg   = (bf16_t*)carve((size_t)NN * FF * 2);
    float*  pad0 = (float*)carve(256);     // zeroed guard (Xg K over-read)
    bf16_t* G1   = (bf16_t*)carve((size_t)NN * FF * 2);
    float*  pad1 = (float*)carve(256);
    bf16_t* G2   = (bf16_t*)carve((size_t)NN * FF * 2);
    float*  pad2 = (float*)carve(256);
    float*  ZR   = (float*)carve((size_t)MROWS * 128 * 4);
    float*  HC   = (float*)carve((size_t)MROWS * 64 * 4);
    float*  Hst  = (float*)carve((size_t)MROWS * 64 * 4);
    float*  M2T  = (float*)carve((size_t)NN * (BB * 10) * 4);
    float*  GO   = (float*)carve((size_t)MROWS * 4);

    auto gg = [](int M, int N) { return dim3((unsigned)(N / GBN), (unsigned)(M / GBM)); };
    const dim3 blk(256);

    // weight repack (zero-padded to WKP-row pieces) + state/guard init
    repack_w<<<(3 * WKP * 128 + 255) / 256, blk, 0, stream>>>(egw, 65, 128, WGE);
    repack_w<<<(3 * WKP * 64 + 255) / 256,  blk, 0, stream>>>(euw, 65, 64,  WUE);
    repack_w<<<(3 * WKP * 128 + 255) / 256, blk, 0, stream>>>(dgw, 66, 128, WGD);
    repack_w<<<(3 * WKP * 64 + 255) / 256,  blk, 0, stream>>>(duw, 66, 64,  WUD);
    zero_f32<<<(MROWS * 64 + 255) / 256, blk, 0, stream>>>(Hst, MROWS * 64);
    zero_f32<<<1, 64, 0, stream>>>(pad0, 64);
    zero_f32<<<1, 64, 0, stream>>>(pad1, 64);
    zero_f32<<<1, 64, 0, stream>>>(pad2, 64);

    // encoder supports: S and T2 = 2*S@S - I
    row_softmax<<<NN, blk, 0, stream>>>(emb, 10, S);
    gemm_bf16<5><<<gg(NN, NN), blk, 0, stream>>>(S, NN, S, NN, nullptr, T2, nullptr, NN, NN);

    // -------- encoder: 12 GRU steps --------
    for (int t = 0; t < TT; ++t) {
        // gate input X = [x_t, h, 0pad]
        build_x_enc<<<MROWS / 256, blk, 0, stream>>>(x, t, Hst, ZR, 0, Xg);
        gemm_bf16<4><<<gg(NN, FF), blk, 0, stream>>>(S,  NN, Xg, FF, nullptr, G1, nullptr, FF, NN);
        gemm_bf16<4><<<gg(NN, FF), blk, 0, stream>>>(T2, NN, Xg, FF, nullptr, G2, nullptr, FF, NN);
        // zr = sigmoid(X@W0 + G1@W1 + G2@W2 + b)   (K=WKP, zero weight rows kill A over-read)
        gemm_bf16<0><<<gg(MROWS, 128), blk, 0, stream>>>(Xg, CP, WGE,                 128, ZR, nullptr, nullptr, 128, WKP);
        gemm_bf16<1><<<gg(MROWS, 128), blk, 0, stream>>>(G1, CP, WGE + 1 * WKP * 128, 128, ZR, nullptr, nullptr, 128, WKP);
        gemm_bf16<2><<<gg(MROWS, 128), blk, 0, stream>>>(G2, CP, WGE + 2 * WKP * 128, 128, ZR, nullptr, egb,     128, WKP);
        // update input X2 = [x_t, z*h, 0pad]
        build_x_enc<<<MROWS / 256, blk, 0, stream>>>(x, t, Hst, ZR, 1, Xg);
        gemm_bf16<4><<<gg(NN, FF), blk, 0, stream>>>(S,  NN, Xg, FF, nullptr, G1, nullptr, FF, NN);
        gemm_bf16<4><<<gg(NN, FF), blk, 0, stream>>>(T2, NN, Xg, FF, nullptr, G2, nullptr, FF, NN);
        gemm_bf16<0><<<gg(MROWS, 64), blk, 0, stream>>>(Xg, CP, WUE,                64, HC, nullptr, nullptr, 64, WKP);
        gemm_bf16<1><<<gg(MROWS, 64), blk, 0, stream>>>(G1, CP, WUE + 1 * WKP * 64, 64, HC, nullptr, nullptr, 64, WKP);
        gemm_bf16<3><<<gg(MROWS, 64), blk, 0, stream>>>(G2, CP, WUE + 2 * WKP * 64, 64, HC, nullptr, eub,     64, WKP);
        gru_combine<<<MROWS * 64 / 256, blk, 0, stream>>>(Hst, ZR, HC);
    }

    // -------- decoder supports from adaptive embeddings --------
    dec_emb_kernel<<<MROWS / 256, blk, 0, stream>>>(Hst, FCE, M2T);
    row_softmax<<<NN, blk, 0, stream>>>(M2T, BB * 10, S);
    gemm_bf16<5><<<gg(NN, NN), blk, 0, stream>>>(S, NN, S, NN, nullptr, T2, nullptr, NN, NN);
    go_init<<<MROWS / 256, blk, 0, stream>>>(x, GO);

    // -------- decoder: 12 GRU steps + projection --------
    for (int t = 0; t < TT; ++t) {
        build_x_dec<<<MROWS / 256, blk, 0, stream>>>(GO, ycov, t, Hst, ZR, 0, Xg);
        gemm_bf16<4><<<gg(NN, FF), blk, 0, stream>>>(S,  NN, Xg, FF, nullptr, G1, nullptr, FF, NN);
        gemm_bf16<4><<<gg(NN, FF), blk, 0, stream>>>(T2, NN, Xg, FF, nullptr, G2, nullptr, FF, NN);
        gemm_bf16<0><<<gg(MROWS, 128), blk, 0, stream>>>(Xg, CP, WGD,                 128, ZR, nullptr, nullptr, 128, WKP);
        gemm_bf16<1><<<gg(MROWS, 128), blk, 0, stream>>>(G1, CP, WGD + 1 * WKP * 128, 128, ZR, nullptr, nullptr, 128, WKP);
        gemm_bf16<2><<<gg(MROWS, 128), blk, 0, stream>>>(G2, CP, WGD + 2 * WKP * 128, 128, ZR, nullptr, dgb,     128, WKP);
        build_x_dec<<<MROWS / 256, blk, 0, stream>>>(GO, ycov, t, Hst, ZR, 1, Xg);
        gemm_bf16<4><<<gg(NN, FF), blk, 0, stream>>>(S,  NN, Xg, FF, nullptr, G1, nullptr, FF, NN);
        gemm_bf16<4><<<gg(NN, FF), blk, 0, stream>>>(T2, NN, Xg, FF, nullptr, G2, nullptr, FF, NN);
        gemm_bf16<0><<<gg(MROWS, 64), blk, 0, stream>>>(Xg, CP, WUD,                64, HC, nullptr, nullptr, 64, WKP);
        gemm_bf16<1><<<gg(MROWS, 64), blk, 0, stream>>>(G1, CP, WUD + 1 * WKP * 64, 64, HC, nullptr, nullptr, 64, WKP);
        gemm_bf16<3><<<gg(MROWS, 64), blk, 0, stream>>>(G2, CP, WUD + 2 * WKP * 64, 64, HC, nullptr, dub,     64, WKP);
        gru_combine<<<MROWS * 64 / 256, blk, 0, stream>>>(Hst, ZR, HC);
        proj_kernel<<<MROWS / 256, blk, 0, stream>>>(Hst, pw, pb, GO, out, t);
    }
}
